// MultiHeadedAttention_8117488189620
// MI455X (gfx1250) — matrix operand order repaired and checked by reading, compiled, not run on hardware
//
#include <hip/hip_runtime.h>
#include <hip/hip_bf16.h>
#include <math.h>

// Problem constants (match reference)
#define B_   2
#define S_   2048
#define HID_ 1024
#define H_   16
#define D_   64

typedef __attribute__((ext_vector_type(16))) __bf16         v16bf;
typedef __attribute__((ext_vector_type(8)))  float          v8f;
typedef __attribute__((ext_vector_type(8)))  unsigned short u16x8;
typedef __attribute__((ext_vector_type(16))) unsigned short u16x16;
typedef __attribute__((ext_vector_type(4)))  unsigned short u16x4;
typedef __attribute__((ext_vector_type(4)))  float          f32x4;

static __device__ __forceinline__ unsigned short f2bf(float f) {
  unsigned int u = __float_as_uint(f);
  u += 0x7fffu + ((u >> 16) & 1u);           // round-to-nearest-even
  return (unsigned short)(u >> 16);
}

static __device__ __forceinline__ v8f wmma_bf16(v16bf a, v16bf b, v8f c) {
  return __builtin_amdgcn_wmma_f32_16x16x32_bf16(false, a, false, b, (short)0, c,
                                                 false, false);
}

// A fragment (16x32 bf16): lane = hi*16+lo holds row M=lo;
// elements 0..7 -> K = hi*8+j, elements 8..15 -> K = 16+hi*8+j.
static __device__ __forceinline__ v16bf load_frag_a(const unsigned short* rowK, int hi) {
  u16x8 c0 = *reinterpret_cast<const u16x8*>(rowK + hi * 8);
  u16x8 c1 = *reinterpret_cast<const u16x8*>(rowK + 16 + hi * 8);
  u16x16 r;
#pragma unroll
  for (int i = 0; i < 8; ++i) { r[i] = c0[i]; r[i + 8] = c1[i]; }
  return __builtin_bit_cast(v16bf, r);
}

// B fragment (32x16 bf16): lane holds column N=lo;
// elements 0..7 -> K = hi*8+j, elements 8..15 -> K = 16+hi*8+j.
static __device__ __forceinline__ v16bf load_frag_b(const unsigned short* colK, int hi) {
  u16x8 c0 = *reinterpret_cast<const u16x8*>(colK + hi * 8);
  u16x8 c1 = *reinterpret_cast<const u16x8*>(colK + 16 + hi * 8);
  u16x16 r;
#pragma unroll
  for (int i = 0; i < 8; ++i) { r[i] = c0[i]; r[i + 8] = c1[i]; }
  return __builtin_bit_cast(v16bf, r);
}

// T5 relative-position bucket bias (only lower-triangular side matters pre-mask)
static __device__ __forceinline__ float bias_val(const float* __restrict__ rb,
                                                 int qpos, int kpos, int hh) {
  int rp = qpos - kpos; if (rp < 0) rp = 0;
  int bucket;
  if (rp < 16) bucket = rp;
  else {
    const float inv_log8 = 16.0f / 2.0794415416798357f;  // 16 / ln(128/16)
    bucket = 16 + (int)(__logf((float)rp * 0.0625f) * inv_log8);
    if (bucket > 31) bucket = 31;
  }
  return rb[bucket * 16 + hh];
}

// ---------------- prep: fp32 -> bf16 (x) ----------------
__global__ void __launch_bounds__(256) cvt_x_kernel(const float* __restrict__ x,
                                                    unsigned short* __restrict__ xbf) {
  const size_t t = (size_t)blockIdx.x * 256 + threadIdx.x;   // 1M threads, 4 elems each
  f32x4 v = *reinterpret_cast<const f32x4*>(x + t * 4);
  u16x4 o;
  o[0] = f2bf(v[0]); o[1] = f2bf(v[1]); o[2] = f2bf(v[2]); o[3] = f2bf(v[3]);
  *reinterpret_cast<u16x4*>(xbf + t * 4) = o;
}

// ---------------- prep: coalesced 32x32 tile transpose + cast of weights ----------
__global__ void __launch_bounds__(256) cvt_w_kernel(const float* __restrict__ Wq,
                                                    const float* __restrict__ Wk,
                                                    const float* __restrict__ Wv,
                                                    const float* __restrict__ Wo,
                                                    unsigned short* __restrict__ wqkvT,
                                                    unsigned short* __restrict__ woT) {
  __shared__ float tile[32][33];
  const int tx = threadIdx.x & 31;
  const int tg = threadIdx.x >> 5;             // 8 groups of 4 rows
  const int blk = blockIdx.x;                  // 0..4095
  const int w   = blk >> 10;                   // weight index 0..3
  const int rem = blk & 1023;
  const int n0  = (rem & 31) * 32;
  const int k0  = (rem >> 5) * 32;
  const float* src = (w == 0) ? Wq : (w == 1) ? Wk : (w == 2) ? Wv : Wo;
  unsigned short* dst = (w < 3) ? (wqkvT + (size_t)w * (1024u * 1024u)) : woT;
#pragma unroll
  for (int j = 0; j < 4; ++j) {
    const int row = tg * 4 + j;                // kk offset within tile
    tile[row][tx] = src[(size_t)(k0 + row) * 1024 + n0 + tx];   // coalesced read
  }
  __syncthreads();
#pragma unroll
  for (int j = 0; j < 4; ++j) {
    const int row = tg * 4 + j;                // n offset within tile
    dst[(size_t)(n0 + row) * 1024 + k0 + tx] = f2bf(tile[tx][row]);  // coalesced write
  }
}

// ---------------- fused QKV projection + RoPE ----------------
// One wave -> 16 rows x 64 cols (one head) so the RoPE partner (d +/- 32) is
// accumulator chunk c^2 in the same lane.
__global__ void __launch_bounds__(256) qkv_rope_kernel(
    const unsigned short* __restrict__ xbf,     // [4096][1024]
    const unsigned short* __restrict__ wqkvT,   // [3072][1024] (q|k|v transposed)
    const float* __restrict__ cosT,             // [2048][64]
    const float* __restrict__ sinT,             // [2048][64]
    unsigned short* __restrict__ qout,          // [B][H][S][D]
    unsigned short* __restrict__ kout,          // [B][H][S][D]
    unsigned short* __restrict__ vTout) {       // [B][H][D][S]
  const int lane = threadIdx.x & 31;
  const int wave = threadIdx.x >> 5;
  const int hi = lane >> 4, lo = lane & 15;
  const int task = blockIdx.x * 8 + wave;       // 0..12287
  const int rt = task / 48;                     // 256 row tiles of 16
  const int ct = task % 48;                     // 48 col tiles of 64
  const int row0 = rt * 16;
  const int col0 = ct * 64;

  const unsigned short* arow = xbf + (size_t)(row0 + lo) * 1024;
  const unsigned short* bp[4];
#pragma unroll
  for (int c = 0; c < 4; ++c) bp[c] = wqkvT + (size_t)(col0 + c * 16 + lo) * 1024;

  v8f acc[4] = {};
  for (int kk = 0; kk < 1024; kk += 32) {
    __builtin_prefetch(arow + kk + 256, 0, 1);
    v16bf a = load_frag_a(arow + kk, hi);
#pragma unroll
    for (int c = 0; c < 4; ++c)
      acc[c] = wmma_bf16(a, load_frag_b(bp[c] + kk, hi), acc[c]);
  }

  const int seg = ct >> 4;   // 0=q, 1=k, 2=v
  const int h   = ct & 15;
  if (seg < 2) {
    unsigned short* dst = (seg == 0) ? qout : kout;
#pragma unroll
    for (int r = 0; r < 8; ++r) {
      const int row = row0 + hi * 8 + r;
      const int b = row >> 11;                  // /S
      const int s = row & 2047;
      const float* cr = cosT + (size_t)s * 64;
      const float* sr = sinT + (size_t)s * 64;
#pragma unroll
      for (int c = 0; c < 4; ++c) {
        const int d = c * 16 + lo;
        const float t   = acc[c][r];
        const float tp  = acc[c ^ 2][r];        // partner at d +/- 32
        const float rot = (c < 2) ? -tp : tp;
        const float ov  = t * cr[d] + rot * sr[d];
        dst[((size_t)(b * 16 + h) * 2048 + s) * 64 + d] = f2bf(ov);
      }
    }
  } else {
#pragma unroll
    for (int r = 0; r < 8; ++r) {
      const int row = row0 + hi * 8 + r;
      const int b = row >> 11;
      const int s = row & 2047;
#pragma unroll
      for (int c = 0; c < 4; ++c) {
        const int d = c * 16 + lo;
        vTout[((size_t)(b * 16 + h) * 64 + d) * 2048 + s] = f2bf(acc[c][r]);
      }
    }
  }
}

// ---------------- flash attention: bias + causal + online softmax ----------------
// k-tile = 32 so both QK^T and P*V WMMAs run with full K=32.
__global__ void __launch_bounds__(256) attn_kernel(
    const unsigned short* __restrict__ qbf,   // [B*H][S][D]
    const unsigned short* __restrict__ kbf,   // [B*H][S][D]
    const unsigned short* __restrict__ vT,    // [B*H][D][S]
    const float* __restrict__ rel_bias,       // [32][16]
    unsigned short* __restrict__ ctx) {       // [B][S][H*D]
  __shared__ unsigned short ldsP[8 * 512];    // per-wave 16x32 bf16 P tile
  const int lane = threadIdx.x & 31;
  const int wave = threadIdx.x >> 5;
  const int hi = lane >> 4, lo = lane & 15;
  const int task = blockIdx.x * 8 + wave;     // 0..4095
  const int qt = task & 127;
  const int bh = task >> 7;                   // 0..31
  const int q0 = qt * 16;
  const int hh = bh & 15;

  const unsigned short* qrow = qbf + ((size_t)bh * S_ + q0 + lo) * D_;
  const v16bf aq0 = load_frag_a(qrow, hi);          // K(d)=0..31
  const v16bf aq1 = load_frag_a(qrow + 32, hi);     // K(d)=32..63

  float m[8], l[8];
  v8f acc[4] = {};
#pragma unroll
  for (int r = 0; r < 8; ++r) { m[r] = -1e30f; l[r] = 0.0f; }

  unsigned short* myP = ldsP + wave * 512;
  const int ktmax = qt >> 1;                  // 32-wide k tiles; last one ends <= 2047

  for (int kt = 0; kt <= ktmax; ++kt) {
    const int k0 = kt * 32;
    const unsigned short* kc0 = kbf + ((size_t)bh * S_ + k0 + lo) * D_;
    const unsigned short* kc1 = kc0 + 16 * D_;
    v8f sc0 = {}, sc1 = {};
    sc0 = wmma_bf16(aq0, load_frag_b(kc0, hi), sc0);       // N = k0..k0+15
    sc0 = wmma_bf16(aq1, load_frag_b(kc0 + 32, hi), sc0);
    sc1 = wmma_bf16(aq0, load_frag_b(kc1, hi), sc1);       // N = k0+16..k0+31
    sc1 = wmma_bf16(aq1, load_frag_b(kc1 + 32, hi), sc1);

    float corr[8];
#pragma unroll
    for (int r = 0; r < 8; ++r) {
      const int qpos = q0 + hi * 8 + r;
      const int kp0 = k0 + lo;
      const int kp1 = kp0 + 16;
      float v0 = sc0[r] * 0.125f + bias_val(rel_bias, qpos, kp0, hh);
      float v1 = sc1[r] * 0.125f + bias_val(rel_bias, qpos, kp1, hh);
      if (kp0 > qpos) v0 = -1.0e9f;            // causal
      if (kp1 > qpos) v1 = -1.0e9f;
      float vmax = fmaxf(v0, v1);              // 32 cols -> local pair + 16-lane reduce
      vmax = fmaxf(vmax, __shfl_xor(vmax, 1, 32));
      vmax = fmaxf(vmax, __shfl_xor(vmax, 2, 32));
      vmax = fmaxf(vmax, __shfl_xor(vmax, 4, 32));
      vmax = fmaxf(vmax, __shfl_xor(vmax, 8, 32));
      const float mn = fmaxf(m[r], vmax);
      const float pe0 = __expf(v0 - mn);
      const float pe1 = __expf(v1 - mn);
      float rs = pe0 + pe1;
      rs += __shfl_xor(rs, 1, 32);
      rs += __shfl_xor(rs, 2, 32);
      rs += __shfl_xor(rs, 4, 32);
      rs += __shfl_xor(rs, 8, 32);
      corr[r] = __expf(m[r] - mn);
      l[r] = l[r] * corr[r] + rs;
      m[r] = mn;
      myP[(hi * 8 + r) * 32 + lo]      = f2bf(pe0);  // C-layout -> LDS row-major 16x32
      myP[(hi * 8 + r) * 32 + 16 + lo] = f2bf(pe1);
    }
#pragma unroll
    for (int c = 0; c < 4; ++c)
#pragma unroll
      for (int r = 0; r < 8; ++r) acc[c][r] *= corr[r];

    asm volatile("s_wait_dscnt 0" ::: "memory"); // wave-private LDS: stores -> loads

    // P (16x32) as full-K A-fragment straight from LDS
    const v16bf ap = load_frag_a(myP + lo * 32, hi);

#pragma unroll
    for (int c = 0; c < 4; ++c) {
      const unsigned short* vp = vT + ((size_t)bh * D_ + c * 16 + lo) * S_ + k0;
      acc[c] = wmma_bf16(ap, load_frag_b(vp, hi), acc[c]);   // full K=32
    }
  }

#pragma unroll
  for (int r = 0; r < 8; ++r) {
    const float invl = 1.0f / l[r];
    const int srow = q0 + hi * 8 + r;
    const int b = bh >> 4;
    unsigned short* crow = ctx + ((size_t)b * S_ + srow) * (H_ * D_) + hh * D_;
#pragma unroll
    for (int c = 0; c < 4; ++c)
      crow[c * 16 + lo] = f2bf(acc[c][r] * invl);
  }
}

// ---------------- output projection: ctx(bf16) @ Wo^T(bf16) -> f32 ----------------
__global__ void __launch_bounds__(256) out_gemm_kernel(
    const unsigned short* __restrict__ ctxbf,   // [4096][1024]
    const unsigned short* __restrict__ woT,     // [1024][1024] transposed
    float* __restrict__ out) {                  // [4096][1024] f32
  const int lane = threadIdx.x & 31;
  const int wave = threadIdx.x >> 5;
  const int hi = lane >> 4, lo = lane & 15;
  const int task = blockIdx.x * 8 + wave;       // 0..4095
  const int rt = task >> 4;
  const int ct = task & 15;
  const int row0 = rt * 16, col0 = ct * 64;

  const unsigned short* arow = ctxbf + (size_t)(row0 + lo) * 1024;
  const unsigned short* bp[4];
#pragma unroll
  for (int c = 0; c < 4; ++c) bp[c] = woT + (size_t)(col0 + c * 16 + lo) * 1024;

  v8f acc[4] = {};
  for (int kk = 0; kk < 1024; kk += 32) {
    __builtin_prefetch(arow + kk + 256, 0, 1);
    v16bf a = load_frag_a(arow + kk, hi);
#pragma unroll
    for (int c = 0; c < 4; ++c)
      acc[c] = wmma_bf16(a, load_frag_b(bp[c] + kk, hi), acc[c]);
  }
#pragma unroll
  for (int r = 0; r < 8; ++r) {
    float* orow = out + (size_t)(row0 + hi * 8 + r) * 1024 + col0;
#pragma unroll
    for (int c = 0; c < 4; ++c) orow[c * 16 + lo] = acc[c][r];
  }
}

// ---------------- host launch ----------------
extern "C" void kernel_launch(void* const* d_in, const int* in_sizes, int n_in,
                              void* d_out, int out_size, void* d_ws, size_t ws_size,
                              hipStream_t stream) {
  const float* x  = (const float*)d_in[0];
  const float* Wq = (const float*)d_in[1];
  const float* Wk = (const float*)d_in[2];
  const float* Wv = (const float*)d_in[3];
  const float* Wo = (const float*)d_in[4];
  const float* rb = (const float*)d_in[5];
  const float* cs = (const float*)d_in[6];
  const float* sn = (const float*)d_in[7];
  float* out = (float*)d_out;

  char* ws = (char*)d_ws;
  unsigned short* xbf   = (unsigned short*)(ws);              //  8.0 MB
  unsigned short* wqkvT = (unsigned short*)(ws + 8388608);    //  6.0 MB
  unsigned short* woT   = (unsigned short*)(ws + 14680064);   //  2.0 MB
  unsigned short* qbf   = (unsigned short*)(ws + 16777216);   //  8.0 MB
  unsigned short* kbf   = (unsigned short*)(ws + 25165824);   //  8.0 MB
  unsigned short* vT    = (unsigned short*)(ws + 33554432);   //  8.0 MB
  unsigned short* ctx   = (unsigned short*)(ws + 41943040);   //  8.0 MB  (total ~50 MB)

  cvt_x_kernel   <<<4096, 256, 0, stream>>>(x, xbf);
  cvt_w_kernel   <<<4096, 256, 0, stream>>>(Wq, Wk, Wv, Wo, wqkvT, woT);
  qkv_rope_kernel<<<1536, 256, 0, stream>>>(xbf, wqkvT, cs, sn, qbf, kbf, vT);
  attn_kernel    <<<512,  256, 0, stream>>>(qbf, kbf, vT, rb, ctx);
  out_gemm_kernel<<<512,  256, 0, stream>>>(ctx, woT, out);

  (void)in_sizes; (void)n_in; (void)out_size; (void)ws_size;
}
